// LinearAttention_13374528159817
// MI455X (gfx1250) — matrix operand order repaired and checked by reading, not compiled, not run on hardware
//
#include <hip/hip_runtime.h>
#include <hip/hip_bf16.h>

#define HEADS 16
#define DIM_HEAD 64
#define SCALE 0.125f   // 64^-0.5
#define BQ 4
#define NQ 4096
#define DQ 1024
#define INNER 1024
#define ROWS (BQ*NQ)        // 16384
#define QKVC (3*INNER)      // 3072

typedef __bf16 bf16;
typedef bf16  v16bf __attribute__((ext_vector_type(16)));
typedef float v8f   __attribute__((ext_vector_type(8)));

union FragAB { v16bf v; float4 f[2]; };

// Async global -> LDS copy of 16 bytes per lane (CDNA5, ASYNCcnt-tracked).
__device__ __forceinline__ void async_copy_b128(unsigned lds_addr, const bf16* gaddr) {
    asm volatile("global_load_async_to_lds_b128 %0, %1, off"
                 :: "v"(lds_addr), "v"(gaddr) : "memory");
}

// ---------------------------------------------------------------------------
// fp32 -> bf16 grid-stride convert
// ---------------------------------------------------------------------------
__global__ __launch_bounds__(256) void la_cvt_bf16(const float* __restrict__ src,
                                                   bf16* __restrict__ dst, int n) {
    for (int i = blockIdx.x * blockDim.x + threadIdx.x; i < n; i += gridDim.x * blockDim.x)
        dst[i] = (bf16)src[i];
}

// ---------------------------------------------------------------------------
// fp32 [R][C] -> bf16 [C][R]  (tiled transpose-convert, LDS 32x33)
// ---------------------------------------------------------------------------
__global__ __launch_bounds__(256) void la_transpose_cvt(const float* __restrict__ src,
                                                        bf16* __restrict__ dst,
                                                        int R, int C) {
    __shared__ float tile[32][33];
    const int tx = threadIdx.x & 31, ty = threadIdx.x >> 5;   // 32 x 8
    const int c0 = blockIdx.x * 32, r0 = blockIdx.y * 32;
    #pragma unroll
    for (int j = 0; j < 32; j += 8)
        tile[ty + j][tx] = src[(size_t)(r0 + ty + j) * C + c0 + tx];
    __syncthreads();
    #pragma unroll
    for (int j = 0; j < 32; j += 8)
        dst[(size_t)(c0 + ty + j) * R + r0 + tx] = (bf16)tile[tx][ty + j];
}

// ---------------------------------------------------------------------------
// bf16 WMMA GEMM: C[M,N] = A[M,K] * B^T[N,K] (+bias).
// Block = 8 waves (2x4), block tile 128x256, K-step 32.
// A/B tiles staged in LDS via async global->LDS copies, double-buffered;
// fragments fed by ds_load_b128; 16 v_wmma_f32_16x16x32_bf16 per wave/K-step.
// Steady-state loop is branch-free; last K-step peeled for the asynccnt drain.
// M%128==0, N%256==0, K%32==0 (true here).
// ---------------------------------------------------------------------------
__global__ __launch_bounds__(256) void la_gemm_bf16_wmma(
    const bf16* __restrict__ A, const bf16* __restrict__ BT,
    const float* __restrict__ bias, float* __restrict__ C,
    int M, int N, int K) {
    __shared__ __align__(16) bf16 sA[2][128][32];   // 16 KB
    __shared__ __align__(16) bf16 sB[2][256][32];   // 32 KB

    const int t    = threadIdx.x;
    const int lane = t & 31;
    const int wave = t >> 5;
    const int wm = wave >> 2;          // 0..1
    const int wn = wave & 3;           // 0..3
    const int l16 = lane & 15;
    const int hi  = lane >> 4;         // 0/1
    const int tileN = blockIdx.x * 256 + wn * 64;
    const int tileM = blockIdx.y * 128 + wm * 64;

    // Global->LDS staging pattern: thread t moves b128 chunk (row = t>>2,
    // quad = t&3) plus row+64 for A, and rows +0/64/128/192 for B.
    const int r4 = t >> 2, q4 = t & 3;
    const bf16* aG = A  + (size_t)(blockIdx.y * 128 + r4) * K + q4 * 8;
    const bf16* bG = BT + (size_t)(blockIdx.x * 256 + r4) * K + q4 * 8;
    const unsigned aL0 = (unsigned)(unsigned long long)&sA[0][r4][q4 * 8];
    const unsigned bL0 = (unsigned)(unsigned long long)&sB[0][r4][q4 * 8];

    auto prefetch = [&](int buf, int k0) {
        const unsigned aL = aL0 + (unsigned)buf * (128 * 32 * 2);
        const unsigned bL = bL0 + (unsigned)buf * (256 * 32 * 2);
        async_copy_b128(aL,                 aG + k0);                       // 6 async
        async_copy_b128(aL +  64 * 32 * 2,  aG + (size_t)64  * K + k0);    // instrs
        async_copy_b128(bL,                 bG + k0);                       // per wave
        async_copy_b128(bL +  64 * 32 * 2,  bG + (size_t)64  * K + k0);    // per stage
        async_copy_b128(bL + 128 * 32 * 2,  bG + (size_t)128 * K + k0);
        async_copy_b128(bL + 192 * 32 * 2,  bG + (size_t)192 * K + k0);
    };

    v8f acc[4][4] = {};

    auto compute = [&](int buf) {
        const bf16 (*As)[32] = sA[buf];
        const bf16 (*Bs)[32] = sB[buf];
        FragAB a[4], b[4];
        #pragma unroll
        for (int i = 0; i < 4; ++i) {
            // A frag (16x32): lanes 0-15 K={0..7,16..23}, lanes 16-31 +8.
            const float4* pa = reinterpret_cast<const float4*>(&As[wm * 64 + i * 16 + l16][hi * 8]);
            a[i].f[0] = pa[0];
            a[i].f[1] = pa[2];
            // B frag (32x16) from BT rows: lane = column, K={0..7,16..23}, lanes 16-31 +8.
            const float4* pb = reinterpret_cast<const float4*>(&Bs[wn * 64 + i * 16 + l16][hi * 8]);
            b[i].f[0] = pb[0];
            b[i].f[1] = pb[2];
        }
        #pragma unroll
        for (int i = 0; i < 4; ++i)
            #pragma unroll
            for (int j = 0; j < 4; ++j)
                acc[i][j] = __builtin_amdgcn_wmma_f32_16x16x32_bf16(
                    false, a[i].v, false, b[j].v, (short)0, acc[i][j], false, false);
    };

    const int nk = K >> 5;
    prefetch(0, 0);
    // Steady state: branch-free prefetch/wait/compute with double buffer.
    for (int ks = 0; ks < nk - 1; ++ks) {
        const int buf = ks & 1;
        prefetch(buf ^ 1, (ks + 1) << 5);
        // 12 outstanding: wait until only the just-issued 6 remain.
        asm volatile("s_wait_asynccnt 0x6" ::: "memory");
        __syncthreads();            // stage `ks` visible to all waves
        compute(buf);
        __syncthreads();            // reads done before buf^1 is overwritten
    }
    // Epilogue: drain async counter, consume the final stage.
    asm volatile("s_wait_asynccnt 0x0" ::: "memory");
    __syncthreads();
    compute((nk - 1) & 1);

    // C/D layout: lanes 0-15 col=lane, elem e -> row e; lanes 16-31 row e+8.
    #pragma unroll
    for (int i = 0; i < 4; ++i) {
        #pragma unroll
        for (int j = 0; j < 4; ++j) {
            const int col = tileN + j * 16 + l16;
            const float badd = bias ? bias[col] : 0.0f;
            float* cp = C + (size_t)(tileM + i * 16 + hi * 8) * N + col;
            #pragma unroll
            for (int e = 0; e < 8; ++e)
                cp[(size_t)e * N] = acc[i][j][e] + badd;
        }
    }
}

// ---------------------------------------------------------------------------
// Per (b,h): column max / sum-exp of k over the sequence dim (softmax axis=-2)
// ---------------------------------------------------------------------------
__global__ __launch_bounds__(256) void la_kcol_stats(const float* __restrict__ qkv,
                                                     float* __restrict__ colmax,
                                                     float* __restrict__ colsum) {
    const int bh = blockIdx.x, b = bh >> 4, h = bh & 15;
    const int e = threadIdx.x & 63, g = threadIdx.x >> 6;   // 4 row-groups x 64 cols
    const float* kp = qkv + (size_t)b * NQ * QKVC + INNER + h * 64 + e;
    __shared__ float red[256];
    __shared__ float cmax[64];

    float m = -3.4e38f;
    for (int n = g; n < NQ; n += 4)
        m = fmaxf(m, kp[(size_t)n * QKVC]);
    red[threadIdx.x] = m;
    __syncthreads();
    if (g == 0) {
        m = fmaxf(fmaxf(red[e], red[e + 64]), fmaxf(red[e + 128], red[e + 192]));
        cmax[e] = m;
    }
    __syncthreads();
    m = cmax[e];
    float s = 0.0f;
    for (int n = g; n < NQ; n += 4)
        s += __expf(kp[(size_t)n * QKVC] - m);
    red[threadIdx.x] = s;
    __syncthreads();
    if (g == 0) {
        s = red[e] + red[e + 64] + red[e + 128] + red[e + 192];
        colmax[bh * 64 + e] = m;
        colsum[bh * 64 + e] = s;
    }
}

// ---------------------------------------------------------------------------
// Partial context: ctxp[bh][seg][d][e] = sum_{n in seg} softmax_row(q*SCALE)[n,d]
//                                                      * exp(k[n,e]-colmax[e])
// grid (64, 8): (b,h) x sequence segment of 512 rows; 256 threads.
// ---------------------------------------------------------------------------
__global__ __launch_bounds__(256) void la_context_partial(const float* __restrict__ qkv,
                                                          const float* __restrict__ colmax,
                                                          float* __restrict__ ctxp) {
    const int bh = blockIdx.x, seg = blockIdx.y;
    const int b = bh >> 4, h = bh & 15;
    const int t = threadIdx.x;
    __shared__ float sq[16][64];
    __shared__ float sk[16][64];
    __shared__ float cm[64];
    if (t < 64) cm[t] = colmax[bh * 64 + t];

    float acc[16];
    #pragma unroll
    for (int j = 0; j < 16; ++j) acc[j] = 0.0f;
    const int d = t >> 2, e0 = (t & 3) << 4;

    for (int c = 0; c < 32; ++c) {                  // 32 chunks * 16 rows = 512
        const int n0 = seg * 512 + c * 16;
        __syncthreads();                            // protect LDS reuse
        #pragma unroll
        for (int i = 0; i < 4; ++i) {
            const int idx = t + i * 256, r = idx >> 6, dd = idx & 63;
            const float* base = qkv + (size_t)(b * NQ + n0 + r) * QKVC + h * 64 + dd;
            sq[r][dd] = base[0] * SCALE;              // q
            sk[r][dd] = __expf(base[INNER] - cm[dd]); // exp(k - colmax)
        }
        __syncthreads();
        if (t < 16) {                               // row softmax of q (64 wide)
            const int r = t;
            float m = -3.4e38f;
            for (int j = 0; j < 64; ++j) m = fmaxf(m, sq[r][j]);
            float s = 0.0f;
            for (int j = 0; j < 64; ++j) { float ev = __expf(sq[r][j] - m); sq[r][j] = ev; s += ev; }
            const float inv = 1.0f / s;
            for (int j = 0; j < 64; ++j) sq[r][j] *= inv;
        }
        __syncthreads();
        #pragma unroll
        for (int r = 0; r < 16; ++r) {
            const float qd = sq[r][d];
            #pragma unroll
            for (int j = 0; j < 16; ++j) acc[j] += qd * sk[r][e0 + j];
        }
    }
    float* outp = ctxp + (((size_t)bh * 8 + seg) * 64 + d) * 64 + e0;
    #pragma unroll
    for (int j = 0; j < 16; ++j) outp[j] = acc[j];
}

// ---------------------------------------------------------------------------
// Reduce 8 segments, apply colsum normalization -> ctx[bh][64][64]
// ---------------------------------------------------------------------------
__global__ __launch_bounds__(256) void la_context_reduce(const float* __restrict__ ctxp,
                                                         const float* __restrict__ colsum,
                                                         float* __restrict__ ctx) {
    const int bh = blockIdx.x;
    for (int idx = threadIdx.x; idx < 4096; idx += 256) {
        float s = 0.0f;
        #pragma unroll
        for (int g = 0; g < 8; ++g) s += ctxp[((size_t)bh * 8 + g) * 4096 + idx];
        ctx[(size_t)bh * 4096 + idx] = s / colsum[bh * 64 + (idx & 63)];
    }
}

// ---------------------------------------------------------------------------
// hout[n, h*64+e] = sum_d v[n,d] * ctx[d,e], written as bf16 (A of GEMM2).
// grid (64, 64): (b,h) x 64-row chunk; 256 threads; LDS stages ctx + v.
// ---------------------------------------------------------------------------
__global__ __launch_bounds__(256) void la_apply_context(const float* __restrict__ qkv,
                                                        const float* __restrict__ ctx,
                                                        bf16* __restrict__ hout) {
    const int bh = blockIdx.x, chunk = blockIdx.y;
    const int b = bh >> 4, h = bh & 15;
    const int t = threadIdx.x;
    __shared__ float sctx[64][64];
    __shared__ float sv[64][64];
    const int n0 = chunk * 64;
    #pragma unroll
    for (int i = 0; i < 16; ++i) {
        const int idx = t + i * 256, r = idx >> 6, dd = idx & 63;
        sctx[r][dd] = ctx[(size_t)bh * 4096 + idx];
        sv[r][dd]   = qkv[(size_t)(b * NQ + n0 + r) * QKVC + 2 * INNER + h * 64 + dd];
    }
    __syncthreads();
    const int r = t >> 2, e0 = (t & 3) << 4;
    float acc[16];
    #pragma unroll
    for (int j = 0; j < 16; ++j) acc[j] = 0.0f;
    for (int dd = 0; dd < 64; ++dd) {
        const float vv = sv[r][dd];
        #pragma unroll
        for (int j = 0; j < 16; ++j) acc[j] += vv * sctx[dd][e0 + j];
    }
    bf16* op = hout + (size_t)(b * NQ + n0 + r) * INNER + h * 64 + e0;
    #pragma unroll
    for (int j = 0; j < 16; ++j) op[j] = (bf16)acc[j];
}

// ---------------------------------------------------------------------------
extern "C" void kernel_launch(void* const* d_in, const int* in_sizes, int n_in,
                              void* d_out, int out_size, void* d_ws, size_t ws_size,
                              hipStream_t stream) {
    const float* x     = (const float*)d_in[0];   // [4,4096,1024]
    const float* w_qkv = (const float*)d_in[1];   // [1024,3072]
    const float* w_out = (const float*)d_in[2];   // [1024,1024]
    const float* b_out = (const float*)d_in[3];   // [1024]
    float* out = (float*)d_out;                   // [4,4096,1024]

    char* ws = (char*)d_ws;
    size_t off = 0;
    auto alloc = [&](size_t bytes) -> void* {
        off = (off + 255) & ~(size_t)255;
        void* p = ws + off;
        off += bytes;
        return p;
    };
    bf16*  xb     = (bf16*) alloc((size_t)ROWS * DQ * sizeof(bf16));       // 32 MB
    bf16*  wqkvT  = (bf16*) alloc((size_t)QKVC * DQ * sizeof(bf16));       // 6 MB
    bf16*  woutT  = (bf16*) alloc((size_t)DQ * INNER * sizeof(bf16));      // 2 MB
    float* qkv    = (float*)alloc((size_t)ROWS * QKVC * sizeof(float));    // 192 MB
    float* cmax   = (float*)alloc((size_t)64 * 64 * sizeof(float));
    float* csum   = (float*)alloc((size_t)64 * 64 * sizeof(float));
    float* ctxp   = (float*)alloc((size_t)64 * 8 * 4096 * sizeof(float));  // 8 MB
    float* ctx    = (float*)alloc((size_t)64 * 4096 * sizeof(float));      // 1 MB
    bf16*  hout   = (bf16*) alloc((size_t)ROWS * INNER * sizeof(bf16));    // 32 MB

    // 1) stage x as bf16
    la_cvt_bf16<<<dim3(4096), dim3(256), 0, stream>>>(x, xb, ROWS * DQ);
    // 2) transpose-convert weights to [N,K] bf16
    la_transpose_cvt<<<dim3(QKVC / 32, DQ / 32), dim3(256), 0, stream>>>(w_qkv, wqkvT, DQ, QKVC);
    la_transpose_cvt<<<dim3(DQ / 32, INNER / 32), dim3(256), 0, stream>>>(w_out, woutT, INNER, DQ);
    // 3) qkv = x @ w_qkv   (WMMA bf16, fp32 accum, async-LDS double buffered)
    la_gemm_bf16_wmma<<<dim3(QKVC / 256, ROWS / 128), dim3(256), 0, stream>>>(
        xb, wqkvT, nullptr, qkv, ROWS, QKVC, DQ);
    // 4) k column softmax stats
    la_kcol_stats<<<dim3(64), dim3(256), 0, stream>>>(qkv, cmax, csum);
    // 5) context = qs^T @ ks (partial over 8 sequence segments, then reduce)
    la_context_partial<<<dim3(64, 8), dim3(256), 0, stream>>>(qkv, cmax, ctxp);
    la_context_reduce<<<dim3(64), dim3(256), 0, stream>>>(ctxp, csum, ctx);
    // 6) hout = V @ ctx  -> bf16
    la_apply_context<<<dim3(64, 64), dim3(256), 0, stream>>>(qkv, ctx, hout);
    // 7) out = hout @ w_out + b_out   (WMMA bf16, fp32 accum)
    la_gemm_bf16_wmma<<<dim3(INNER / 256, ROWS / 128), dim3(256), 0, stream>>>(
        hout, woutT, b_out, out, ROWS, INNER, DQ);
}
